// EGNN_Layer_38448547234250
// MI455X (gfx1250) — matrix operand order repaired and checked by reading, not compiled, not run on hardware
//
#include <hip/hip_runtime.h>
#include <stdint.h>

// ---------------------------------------------------------------------------
// EGNN layer for MI455X (gfx1250): f16 WMMA (f32 accumulate) for all GEMMs,
// atomic f32 scatter for segment sums, wave32-tiled (1 wave = 16-row tile).
// ---------------------------------------------------------------------------

typedef __attribute__((ext_vector_type(16))) _Float16 v16h;
typedef __attribute__((ext_vector_type(8)))  _Float16 v8h;
typedef __attribute__((ext_vector_type(8)))  float    v8f;
typedef _Float16 f16;

#define WPB 4            // waves per block (128 threads)
#define HID 64
#define EDGE_NF 16

// packed-weight offsets (in halves) inside the pack region of d_ws
#define PK_EMW1 0        // [160x64] Kpad=160 (K=146)
#define PK_EMW2 10240    // [64x64]
#define PK_CNW1 14336    // [64x64]
#define PK_NNW1 18432    // [128x64]
#define PK_NNW2 26624    // [64x64]
#define PK_NXW1 30720    // [64x64]
#define PK_NVW1 34816    // [64x64]
#define PK_TOTAL 38912

__device__ __forceinline__ void wave_lds_sync() {
  // intra-wave LDS producer->consumer ordering; waves are independent so no
  // block barrier is needed (allows early wave exit).
  asm volatile("s_wait_dscnt 0" ::: "memory");
  __builtin_amdgcn_wave_barrier();
}

__device__ __forceinline__ float silu(float x) {
  return x / (1.0f + __expf(-x));
}

// A-operand fragment for v_wmma_f32_16x16x32_f16 from a row-major LDS tile.
// lane<16 : row m=lane,   elems 0..7 = K k0..k0+7,  8..15 = K k0+16..k0+23
// lane>=16: row m=lane-16, elems 0..7 = K k0+8..+15, 8..15 = K k0+24..+31
__device__ __forceinline__ v16h load_a_tile(const f16* base, int lane, int k0,
                                            int stride) {
  int m = lane & 15;
  int off = (lane & 16) ? 8 : 0;
  const f16* p = base + m * stride + k0 + off;
  union { v16h v; v8h h[2]; } u;
  u.h[0] = *(const v8h*)(p);
  u.h[1] = *(const v8h*)(p + 16);
  return u.v;
}

__device__ __forceinline__ v8f wmma_f16(v16h a, v16h b, v8f c) {
  return __builtin_amdgcn_wmma_f32_16x16x32_f16(false, a, false, b, (short)0,
                                                c, false, false);
}

// ---------------------------------------------------------------------------
// zero the atomic accumulators
// ---------------------------------------------------------------------------
__global__ void egnn_zero_kernel(float* __restrict__ p, long n) {
  long i = (long)blockIdx.x * blockDim.x + threadIdx.x;
  if (i < n) p[i] = 0.0f;
}

// ---------------------------------------------------------------------------
// pack fp32 [K,64] row-major weights into the B-operand lane layout:
// slot s = ((kc*4 + nt)*32 + lane)*16 + i ;  n = nt*16 + (lane&15)
// k = kc*32 + ((lane&16)?8:0) + (i&7) + ((i&8)?16:0) ; zero-pad k >= K.
// ---------------------------------------------------------------------------
__device__ void pack_one(const float* __restrict__ W, int K, int Kpad,
                         f16* __restrict__ dst, int tid, int nth) {
  int total = (Kpad >> 5) * 4 * 32 * 16;
  for (int s = tid; s < total; s += nth) {
    int i = s & 15;
    int lane = (s >> 4) & 31;
    int nt = (s >> 9) & 3;
    int kc = s >> 11;
    int n = nt * 16 + (lane & 15);
    int k = kc * 32 + ((lane & 16) ? 8 : 0) + (i & 7) + ((i & 8) ? 16 : 0);
    dst[s] = (f16)((k < K) ? W[k * 64 + n] : 0.0f);
  }
}

__global__ void egnn_pack_kernel(const float* __restrict__ em_w1,
                                 const float* __restrict__ em_w2,
                                 const float* __restrict__ cn_w1,
                                 const float* __restrict__ nn_w1,
                                 const float* __restrict__ nn_w2,
                                 const float* __restrict__ nx_w1,
                                 const float* __restrict__ nv_w1,
                                 f16* __restrict__ dst) {
  int tid = blockIdx.x * blockDim.x + threadIdx.x;
  int nth = gridDim.x * blockDim.x;
  pack_one(em_w1, 146, 160, dst + PK_EMW1, tid, nth);
  pack_one(em_w2, 64, 64, dst + PK_EMW2, tid, nth);
  pack_one(cn_w1, 64, 64, dst + PK_CNW1, tid, nth);
  pack_one(nn_w1, 128, 128, dst + PK_NNW1, tid, nth);
  pack_one(nn_w2, 64, 64, dst + PK_NNW2, tid, nth);
  pack_one(nx_w1, 64, 64, dst + PK_NXW1, tid, nth);
  pack_one(nv_w1, 64, 64, dst + PK_NVW1, tid, nth);
}

// ---------------------------------------------------------------------------
// Edge kernel: 1 wave per 16-edge tile. 36 WMMAs / tile.
// features: [ |rij|, |vij|, h[row](64), h[col](64), edge_fea(16), pad(14) ]
// ---------------------------------------------------------------------------
__global__ void __launch_bounds__(32 * WPB)
egnn_edge_kernel(const float* __restrict__ x, const float* __restrict__ h,
                 const float* __restrict__ v, const long long* __restrict__ ei,
                 const float* __restrict__ ef, const f16* __restrict__ packs,
                 const float* __restrict__ em_b1,
                 const float* __restrict__ em_b2,
                 const float* __restrict__ cn_b1,
                 const float* __restrict__ cn_w2,
                 const float* __restrict__ cn_b2, float* __restrict__ cnt,
                 float* __restrict__ totf, float* __restrict__ totm,
                 int nEdge, int nTiles) {
  __shared__ __align__(16) f16   sA[WPB][16 * 160];
  __shared__ __align__(16) f16   sM[WPB][16 * 64];
  __shared__ __align__(16) f16   sM2[WPB][16 * 64];
  __shared__ __align__(16) float sC[WPB][16 * 64];
  __shared__ __align__(16) float sRV[WPB][16 * 4];
  __shared__ __align__(16) int   sIdx[WPB][32];

  const int wave = threadIdx.x >> 5;
  const int lane = threadIdx.x & 31;
  const int tile = blockIdx.x * WPB + wave;
  if (tile >= nTiles) return;  // wave-local control only; safe
  const long e0 = (long)tile * 16;

  f16* A = sA[wave];
  int* idx = sIdx[wave];

  if (lane < 16) {
    idx[lane] = (int)ei[e0 + lane];             // row
    idx[16 + lane] = (int)ei[nEdge + e0 + lane];  // col
  }
  wave_lds_sync();

  // geometry + invariants (lanes 0..15, one edge each)
  if (lane < 16) {
    int r = idx[lane], c = idx[16 + lane];
    float rx = x[r * 3 + 0] - x[c * 3 + 0];
    float ry = x[r * 3 + 1] - x[c * 3 + 1];
    float rz = x[r * 3 + 2] - x[c * 3 + 2];
    float wx = v[r * 3 + 0] - v[c * 3 + 0];
    float wy = v[r * 3 + 1] - v[c * 3 + 1];
    float wz = v[r * 3 + 2] - v[c * 3 + 2];
    float* rv = &sRV[wave][lane * 4];
    rv[0] = rx; rv[1] = ry; rv[2] = rz;
    A[lane * 160 + 0] = (f16)sqrtf(rx * rx + ry * ry + rz * rz);
    A[lane * 160 + 1] = (f16)sqrtf(wx * wx + wy * wy + wz * wz);
#pragma unroll
    for (int i = 146; i < 160; ++i) A[lane * 160 + i] = (f16)0.0f;
  }
  // cooperative gathers: 32 lanes fill one edge row per iteration
  for (int e = 0; e < 16; ++e) {
    int r = idx[e], c = idx[16 + e];
    A[e * 160 + 2 + lane]  = (f16)h[r * 64 + lane];
    A[e * 160 + 34 + lane] = (f16)h[r * 64 + 32 + lane];
    A[e * 160 + 66 + lane] = (f16)h[c * 64 + lane];
    A[e * 160 + 98 + lane] = (f16)h[c * 64 + 32 + lane];
    if (lane < 16) A[e * 160 + 130 + lane] = (f16)ef[(e0 + e) * 16 + lane];
  }
  wave_lds_sync();

  const int nlo = lane & 15;
  const int mhi = (lane >> 4) << 3;

  // ---- edge MLP layer 1: [16,160] x [160,64] ----
  v8f acc[4] = {};
#pragma unroll
  for (int kc = 0; kc < 5; ++kc) {
    v16h a = load_a_tile(A, lane, kc * 32, 160);
#pragma unroll
    for (int nt = 0; nt < 4; ++nt) {
      v16h b = *(const v16h*)(packs + PK_EMW1 + (((kc << 2) + nt) * 32 + lane) * 16);
      acc[nt] = wmma_f16(a, b, acc[nt]);
    }
  }
  f16* M1 = sM[wave];
#pragma unroll
  for (int nt = 0; nt < 4; ++nt) {
    float bb = em_b1[nt * 16 + nlo];
#pragma unroll
    for (int j = 0; j < 8; ++j) {
      M1[(mhi + j) * 64 + nt * 16 + nlo] = (f16)silu(acc[nt][j] + bb);
    }
  }
  wave_lds_sync();

  // ---- edge MLP layer 2: [16,64] x [64,64], silu -> final message ----
  v8f ac2[4] = {};
#pragma unroll
  for (int kc = 0; kc < 2; ++kc) {
    v16h a = load_a_tile(M1, lane, kc * 32, 64);
#pragma unroll
    for (int nt = 0; nt < 4; ++nt) {
      v16h b = *(const v16h*)(packs + PK_EMW2 + (((kc << 2) + nt) * 32 + lane) * 16);
      ac2[nt] = wmma_f16(a, b, ac2[nt]);
    }
  }
  f16* M2 = sM2[wave];
#pragma unroll
  for (int nt = 0; nt < 4; ++nt) {
    float bb = em_b2[nt * 16 + nlo];
#pragma unroll
    for (int j = 0; j < 8; ++j) {
      float y = silu(ac2[nt][j] + bb);
      M2[(mhi + j) * 64 + nt * 16 + nlo] = (f16)y;
      atomicAdd(&totm[(long)idx[mhi + j] * 64 + nt * 16 + nlo], y);
    }
  }
  wave_lds_sync();

  // ---- coord MLP layer 1: [16,64] x [64,64], silu ----
  v8f ac3[4] = {};
#pragma unroll
  for (int kc = 0; kc < 2; ++kc) {
    v16h a = load_a_tile(M2, lane, kc * 32, 64);
#pragma unroll
    for (int nt = 0; nt < 4; ++nt) {
      v16h b = *(const v16h*)(packs + PK_CNW1 + (((kc << 2) + nt) * 32 + lane) * 16);
      ac3[nt] = wmma_f16(a, b, ac3[nt]);
    }
  }
  float* C = sC[wave];
#pragma unroll
  for (int nt = 0; nt < 4; ++nt) {
    float bb = cn_b1[nt * 16 + nlo];
#pragma unroll
    for (int j = 0; j < 8; ++j) {
      C[(mhi + j) * 64 + nt * 16 + nlo] = silu(ac3[nt][j] + bb);
    }
  }
  wave_lds_sync();

  // ---- coord gate dot + scatter of f = rij * gate (fv is dead code) ----
  if (lane < 16) {
    float s = cn_b2[0];
#pragma unroll
    for (int k = 0; k < 64; ++k) s += C[lane * 64 + k] * cn_w2[k];
    int r = idx[lane];
    const float* rv = &sRV[wave][lane * 4];
    atomicAdd(&totf[r * 3 + 0], rv[0] * s);
    atomicAdd(&totf[r * 3 + 1], rv[1] * s);
    atomicAdd(&totf[r * 3 + 2], rv[2] * s);
    atomicAdd(&cnt[r], 1.0f);
  }
}

// ---------------------------------------------------------------------------
// Node kernel: 1 wave per 16-node tile. 40 WMMAs / tile.
// ---------------------------------------------------------------------------
__global__ void __launch_bounds__(32 * WPB)
egnn_node_kernel(const float* __restrict__ x, const float* __restrict__ h,
                 const float* __restrict__ v, const f16* __restrict__ packs,
                 const float* __restrict__ nn_b1,
                 const float* __restrict__ nn_b2,
                 const float* __restrict__ nx_b1,
                 const float* __restrict__ nx_w2,
                 const float* __restrict__ nx_b2,
                 const float* __restrict__ nv_b1,
                 const float* __restrict__ nv_w2,
                 const float* __restrict__ nv_b2,
                 const float* __restrict__ cnt, const float* __restrict__ totf,
                 const float* __restrict__ totm, float* __restrict__ xout,
                 float* __restrict__ vout, float* __restrict__ hout,
                 int nTiles) {
  __shared__ __align__(16) f16   sA[WPB][16 * 128];
  __shared__ __align__(16) f16   sH[WPB][16 * 64];
  __shared__ __align__(16) float sS[WPB][16 * 64];

  const int wave = threadIdx.x >> 5;
  const int lane = threadIdx.x & 31;
  const int tile = blockIdx.x * WPB + wave;
  if (tile >= nTiles) return;  // only wave-local syncs below; safe
  const int n0 = tile * 16;

  f16* A = sA[wave];
  // A tile: cols 0..63 = h, 64..127 = tot_message (contiguous rows)
  for (int e = 0; e < 16; ++e) {
    long nb = (long)(n0 + e) * 64;
    A[e * 128 + lane]       = (f16)h[nb + lane];
    A[e * 128 + 32 + lane]  = (f16)h[nb + 32 + lane];
    A[e * 128 + 64 + lane]  = (f16)totm[nb + lane];
    A[e * 128 + 96 + lane]  = (f16)totm[nb + 32 + lane];
  }
  wave_lds_sync();

  const int nlo = lane & 15;
  const int mhi = (lane >> 4) << 3;
  float* S = sS[wave];

  // ---- nv gate: silu(h@w1+b1)@w2+b2 ----
  v8f av[4] = {};
#pragma unroll
  for (int kc = 0; kc < 2; ++kc) {
    v16h a = load_a_tile(A, lane, kc * 32, 128);
#pragma unroll
    for (int nt = 0; nt < 4; ++nt) {
      v16h b = *(const v16h*)(packs + PK_NVW1 + (((kc << 2) + nt) * 32 + lane) * 16);
      av[nt] = wmma_f16(a, b, av[nt]);
    }
  }
#pragma unroll
  for (int nt = 0; nt < 4; ++nt) {
    float bb = nv_b1[nt * 16 + nlo];
#pragma unroll
    for (int j = 0; j < 8; ++j)
      S[(mhi + j) * 64 + nt * 16 + nlo] = silu(av[nt][j] + bb);
  }
  wave_lds_sync();
  float gv = 0.0f;
  if (lane < 16) {
    gv = nv_b2[0];
#pragma unroll
    for (int k = 0; k < 64; ++k) gv += S[lane * 64 + k] * nv_w2[k];
  }
  wave_lds_sync();

  // ---- nx gate ----
  v8f ax[4] = {};
#pragma unroll
  for (int kc = 0; kc < 2; ++kc) {
    v16h a = load_a_tile(A, lane, kc * 32, 128);
#pragma unroll
    for (int nt = 0; nt < 4; ++nt) {
      v16h b = *(const v16h*)(packs + PK_NXW1 + (((kc << 2) + nt) * 32 + lane) * 16);
      ax[nt] = wmma_f16(a, b, ax[nt]);
    }
  }
#pragma unroll
  for (int nt = 0; nt < 4; ++nt) {
    float bb = nx_b1[nt * 16 + nlo];
#pragma unroll
    for (int j = 0; j < 8; ++j)
      S[(mhi + j) * 64 + nt * 16 + nlo] = silu(ax[nt][j] + bb);
  }
  wave_lds_sync();
  float gx = 0.0f;
  if (lane < 16) {
    gx = nx_b2[0];
#pragma unroll
    for (int k = 0; k < 64; ++k) gx += S[lane * 64 + k] * nx_w2[k];
  }

  // ---- h_new layer 1: [16,128] x [128,64], silu ----
  v8f an[4] = {};
#pragma unroll
  for (int kc = 0; kc < 4; ++kc) {
    v16h a = load_a_tile(A, lane, kc * 32, 128);
#pragma unroll
    for (int nt = 0; nt < 4; ++nt) {
      v16h b = *(const v16h*)(packs + PK_NNW1 + (((kc << 2) + nt) * 32 + lane) * 16);
      an[nt] = wmma_f16(a, b, an[nt]);
    }
  }
  f16* Hb = sH[wave];
#pragma unroll
  for (int nt = 0; nt < 4; ++nt) {
    float bb = nn_b1[nt * 16 + nlo];
#pragma unroll
    for (int j = 0; j < 8; ++j)
      Hb[(mhi + j) * 64 + nt * 16 + nlo] = (f16)silu(an[nt][j] + bb);
  }
  wave_lds_sync();

  // ---- h_new layer 2: [16,64] x [64,64] (no activation) ----
  v8f ah[4] = {};
#pragma unroll
  for (int kc = 0; kc < 2; ++kc) {
    v16h a = load_a_tile(Hb, lane, kc * 32, 64);
#pragma unroll
    for (int nt = 0; nt < 4; ++nt) {
      v16h b = *(const v16h*)(packs + PK_NNW2 + (((kc << 2) + nt) * 32 + lane) * 16);
      ah[nt] = wmma_f16(a, b, ah[nt]);
    }
  }
#pragma unroll
  for (int nt = 0; nt < 4; ++nt) {
    float bb = nn_b2[nt * 16 + nlo];
#pragma unroll
    for (int j = 0; j < 8; ++j)
      hout[(long)(n0 + mhi + j) * 64 + nt * 16 + nlo] = ah[nt][j] + bb;
  }

  // ---- x_new / v_new epilogue ----
  if (lane < 16) {
    int n = n0 + lane;
    float dn = cnt[n];
    if (dn < 1.0f) dn = 1.0f;
#pragma unroll
    for (int d = 0; d < 3; ++d) {
      float tf = totf[n * 3 + d] / dn;
      tf = fminf(fmaxf(tf, -100.0f), 100.0f);
      vout[n * 3 + d] = gv * v[n * 3 + d] + tf;
      xout[n * 3 + d] = gx * x[n * 3 + d] + tf;
    }
  }
}

// ---------------------------------------------------------------------------
extern "C" void kernel_launch(void* const* d_in, const int* in_sizes, int n_in,
                              void* d_out, int out_size, void* d_ws,
                              size_t ws_size, hipStream_t stream) {
  const float* x = (const float*)d_in[0];
  const float* h = (const float*)d_in[1];
  const float* v = (const float*)d_in[2];
  const long long* ei = (const long long*)d_in[3];
  const float* ef = (const float*)d_in[4];
  const float* em_w1 = (const float*)d_in[5];
  const float* em_b1 = (const float*)d_in[6];
  const float* em_w2 = (const float*)d_in[7];
  const float* em_b2 = (const float*)d_in[8];
  const float* cn_w1 = (const float*)d_in[9];
  const float* cn_b1 = (const float*)d_in[10];
  const float* cn_w2 = (const float*)d_in[11];
  const float* cn_b2 = (const float*)d_in[12];
  const float* nn_w1 = (const float*)d_in[13];
  const float* nn_b1 = (const float*)d_in[14];
  const float* nn_w2 = (const float*)d_in[15];
  const float* nn_b2 = (const float*)d_in[16];
  const float* nx_w1 = (const float*)d_in[17];
  const float* nx_b1 = (const float*)d_in[18];
  const float* nx_w2 = (const float*)d_in[19];
  const float* nx_b2 = (const float*)d_in[20];
  const float* nv_w1 = (const float*)d_in[21];
  const float* nv_b1 = (const float*)d_in[22];
  const float* nv_w2 = (const float*)d_in[23];
  const float* nv_b2 = (const float*)d_in[24];

  const int nNode = in_sizes[0] / 3;         // 50000
  const int nEdge = in_sizes[4] / EDGE_NF;   // 800000

  // workspace layout
  float* cnt = (float*)d_ws;                       // [nNode]
  float* totf = cnt + nNode;                       // [nNode*3]
  float* totm = totf + (size_t)nNode * 3;          // [nNode*64]
  uintptr_t pa = (uintptr_t)(totm + (size_t)nNode * 64);
  pa = (pa + 31) & ~(uintptr_t)31;
  f16* packs = (f16*)pa;                           // [PK_TOTAL] halves

  float* out = (float*)d_out;
  float* xout = out;                          // [nNode*3]
  float* vout = out + (size_t)nNode * 3;      // [nNode*3]
  float* hout = out + (size_t)nNode * 6;      // [nNode*64]

  const long nz = (long)nNode * (1 + 3 + 64);
  egnn_zero_kernel<<<(int)((nz + 255) / 256), 256, 0, stream>>>(cnt, nz);
  egnn_pack_kernel<<<64, 256, 0, stream>>>(em_w1, em_w2, cn_w1, nn_w1, nn_w2,
                                           nx_w1, nv_w1, packs);

  const int eTiles = nEdge / 16;
  egnn_edge_kernel<<<(eTiles + WPB - 1) / WPB, 32 * WPB, 0, stream>>>(
      x, h, v, ei, ef, packs, em_b1, em_b2, cn_b1, cn_w2, cn_b2, cnt, totf,
      totm, nEdge, eTiles);

  const int nTiles = nNode / 16;
  egnn_node_kernel<<<(nTiles + WPB - 1) / WPB, 32 * WPB, 0, stream>>>(
      x, h, v, packs, nn_b1, nn_b2, nx_b1, nx_w2, nx_b2, nv_b1, nv_w2, nv_b2,
      cnt, totf, totm, xout, vout, hout, nTiles);
}
